// RotatedDetectionLoss_6837587935476
// MI455X (gfx1250) — matrix operand order repaired and checked by reading, not compiled, not run on hardware
//
#include <hip/hip_runtime.h>
#include <math.h>

typedef __attribute__((ext_vector_type(16))) _Float16 v16h_t;
typedef __attribute__((ext_vector_type(8)))  float    v8f_t;

#define NC      15
#define CPAD    16
#define NLOG    91
#define B_      16
#define M_      64
#define EPSA    1e-9f
#define ANG_SCALE 0.017453292519943295f  // pi/2/90

// ---------------------------------------------------------------- helpers
__device__ __forceinline__ float probiou_hd_dev(
    float x1, float y1, float a1, float b1, float c1,
    float x2, float y2, float a2, float b2, float c2) {
  const float eps = 1e-3f;
  float A  = a1 + a2, Bs = b1 + b2, Cs = c1 + c2;
  float dx = x1 - x2, dy = y1 - y2;
  float denom = A * Bs - Cs * Cs + eps;
  float t1 = (A * dy * dy + Bs * dx * dx) / denom * 0.25f;
  float t2 = Cs * (x2 - x1) * (y1 - y2) / denom * 0.5f;
  float det1 = fmaxf(a1 * b1 - c1 * c1, 0.0f);
  float det2 = fmaxf(a2 * b2 - c2 * c2, 0.0f);
  float t3 = 0.5f * logf(denom / (4.0f * sqrtf(det1 * det2) + eps) + eps);
  float bd = fminf(fmaxf(t1 + t2 + t3, eps), 100.0f);
  return sqrtf(1.0f - expf(-bd) + 1e-7f);
}

__device__ __forceinline__ void topk13_insert(float (&t)[13], float v) {
  if (v > t[12]) {
    float x = v;
#pragma unroll
    for (int i = 0; i < 13; ++i) {
      float hi = fmaxf(t[i], x);
      x = fminf(t[i], x);
      t[i] = hi;
    }
  }
}

// ---------------------------------------------------------------- K1: per-anchor prep
__global__ void k1_prep(const float* __restrict__ cls_logits,
                        const float* __restrict__ reg_dist,
                        const float* __restrict__ raw_angles,
                        const float* __restrict__ anchor_pts,
                        const float* __restrict__ stride_t,
                        _Float16* __restrict__ sig16,
                        float* __restrict__ dec,
                        float* __restrict__ angstat,
                        int N) {
  int idx = blockIdx.x * blockDim.x + threadIdx.x;
  if (idx >= B_ * N) return;
  int n = idx % N;

  // angle softmax expectation + log-softmax stats
  const float* ra = raw_angles + (size_t)idx * NLOG;
  float mx = -1e30f;
  for (int i = 0; i < NLOG; ++i) mx = fmaxf(mx, ra[i]);
  float se = 0.0f, dot = 0.0f;
  for (int i = 0; i < NLOG; ++i) {
    float e = expf(ra[i] - mx);
    se += e;
    dot += e * (float)i;
  }
  float ang = (dot / se) * ANG_SCALE;
  angstat[(size_t)idx * 2 + 0] = mx;
  angstat[(size_t)idx * 2 + 1] = logf(se);

  // decode box
  const float* rd = reg_dist + (size_t)idx * 4;
  float lt0 = rd[0], lt1 = rd[1], rb0 = rd[2], rb1 = rd[3];
  float st = stride_t[n];
  float ax = anchor_pts[2 * n], ay = anchor_pts[2 * n + 1];
  float o0 = (rb0 - lt0) * 0.5f, o1 = (rb1 - lt1) * 0.5f;
  float cA = cosf(ang), sA = sinf(ang);
  float x = (o0 * cA - o1 * sA) * st + ax;
  float y = (o0 * sA + o1 * cA) * st + ay;
  float w = (lt0 + rb0) * st, h = (lt1 + rb1) * st;
  float Aq = w * w * (1.0f / 12.0f), Bq = h * h * (1.0f / 12.0f);
  float* dp = dec + (size_t)idx * 8;
  dp[0] = x; dp[1] = y; dp[2] = w; dp[3] = h; dp[4] = ang;
  dp[5] = Aq * cA * cA + Bq * sA * sA;
  dp[6] = Aq * sA * sA + Bq * cA * cA;
  dp[7] = (Aq - Bq) * cA * sA;

  // f16 sigmoid table (padded to 16)
  const float* cl = cls_logits + (size_t)idx * NC;
  for (int c = 0; c < CPAD; ++c) {
    float v = (c < NC) ? (1.0f / (1.0f + expf(-cl[c]))) : 0.0f;
    sig16[(size_t)idx * CPAD + c] = (_Float16)v;
  }
}

// ---------------------------------------------------------------- K0: zero atomic-max bufs
__global__ void k0_zero(unsigned* a, unsigned* b, int n) {
  int i = blockIdx.x * blockDim.x + threadIdx.x;
  if (i < n) { a[i] = 0u; b[i] = 0u; }
}

// ---------------------------------------------------------------- K2: per-(b,m) top-13 threshold
__global__ void k2_topk(const float* __restrict__ dec,
                        const _Float16* __restrict__ sig16,
                        const int* __restrict__ gt_labels,
                        const float* __restrict__ gt_boxes,
                        const float* __restrict__ valid_mask,
                        const float* __restrict__ anchor_pts,
                        float* __restrict__ thresh, int N) {
  __shared__ float lds[256 * 13 + 32 * 13];
  int bm = blockIdx.x;
  int b = bm / M_;
  int tid = threadIdx.x;

  const float* gb = gt_boxes + (size_t)bm * 5;
  float gx = gb[0], gy = gb[1], gw = gb[2], gh = gb[3], ga = gb[4];
  int lbl = gt_labels[bm];
  float vmask = valid_mask[bm];
  float ca = cosf(ga), sa = sinf(ga);
  float Aq = gw * gw * (1.0f / 12.0f), Bq = gh * gh * (1.0f / 12.0f);
  float gva = Aq * ca * ca + Bq * sa * sa;
  float gvb = Aq * sa * sa + Bq * ca * ca;
  float gvc = (Aq - Bq) * ca * sa;

  float t[13];
#pragma unroll
  for (int i = 0; i < 13; ++i) t[i] = 0.0f;

  size_t bN = (size_t)b * N;
  for (int n = tid; n < N; n += 256) {
    float ax = anchor_pts[2 * n], ay = anchor_pts[2 * n + 1];
    float dx = ax - gx, dy = ay - gy;
    float xr = dx * ca + dy * sa;
    float yr = -dx * sa + dy * ca;
    bool cand = (fabsf(xr) < gw * 0.5f) && (fabsf(yr) < gh * 0.5f) && (vmask > 0.0f);
    if (!cand) continue;
    const float4* dp = (const float4*)(dec + (bN + n) * 8);
    float4 d0 = dp[0], d1 = dp[1];
    float hd = probiou_hd_dev(gx, gy, gva, gvb, gvc, d0.x, d0.y, d1.y, d1.z, d1.w);
    float iou = fmaxf(1.0f - hd, 0.0f);
    float i2 = iou * iou, i6 = i2 * i2 * i2;
    float score = (float)sig16[(bN + n) * CPAD + lbl];
    float key = score * i6;
    topk13_insert(t, key);
  }
#pragma unroll
  for (int i = 0; i < 13; ++i) lds[tid * 13 + i] = t[i];
  __syncthreads();

  if (tid < 32) {
    float s[13];
#pragma unroll
    for (int i = 0; i < 13; ++i) s[i] = 0.0f;
    for (int src = tid * 8; src < tid * 8 + 8; ++src)
      for (int i = 0; i < 13; ++i) topk13_insert(s, lds[src * 13 + i]);
#pragma unroll
    for (int i = 0; i < 13; ++i) lds[256 * 13 + tid * 13 + i] = s[i];
  }
  __syncthreads();

  if (tid == 0) {
    float s[13];
#pragma unroll
    for (int i = 0; i < 13; ++i) s[i] = 0.0f;
    for (int src = 0; src < 32; ++src)
      for (int i = 0; i < 13; ++i) topk13_insert(s, lds[256 * 13 + src * 13 + i]);
    thresh[bm] = s[12];
  }
}

// ---------------------------------------------------------------- K3: WMMA assigner (1 wave / 16-anchor tile)
__global__ void __launch_bounds__(32) k3_assign(
    const float* __restrict__ dec,
    const _Float16* __restrict__ sig16,
    const int* __restrict__ gt_labels,
    const float* __restrict__ gt_boxes,
    const float* __restrict__ valid_mask,
    const float* __restrict__ anchor_pts,
    const float* __restrict__ thresh,
    int* __restrict__ tgtI, float* __restrict__ fgA,
    float* __restrict__ alA, float* __restrict__ ioA,
    unsigned* __restrict__ maxAl, unsigned* __restrict__ maxIo, int N) {
  __shared__ float gbox[M_][10];  // gx gy gw gh ca sa va vb vc thr
  __shared__ int   glb[M_];
  __shared__ float gval[M_];

  int tiles = N / 16;
  int b = blockIdx.x / tiles;
  int nt = blockIdx.x % tiles;
  int lane = threadIdx.x;

  for (int g = lane; g < M_; g += 32) {
    const float* gb = gt_boxes + ((size_t)b * M_ + g) * 5;
    float gx = gb[0], gy = gb[1], gw = gb[2], gh = gb[3], ga = gb[4];
    float ca = cosf(ga), sa = sinf(ga);
    float Aq = gw * gw * (1.0f / 12.0f), Bq = gh * gh * (1.0f / 12.0f);
    gbox[g][0] = gx; gbox[g][1] = gy; gbox[g][2] = gw; gbox[g][3] = gh;
    gbox[g][4] = ca; gbox[g][5] = sa;
    gbox[g][6] = Aq * ca * ca + Bq * sa * sa;
    gbox[g][7] = Aq * sa * sa + Bq * ca * ca;
    gbox[g][8] = (Aq - Bq) * ca * sa;
    gbox[g][9] = thresh[b * M_ + g];
    glb[g]  = gt_labels[b * M_ + g];
    gval[g] = valid_mask[b * M_ + g];
  }
  __syncthreads();

  int n = nt * 16 + (lane & 15);
  size_t bN = (size_t)b * N;
  const float4* dp = (const float4*)(dec + (bN + n) * 8);
  float4 d0 = dp[0], d1 = dp[1];
  float px = d0.x, py = d0.y, pva = d1.y, pvb = d1.z, pvc = d1.w;
  float ax = anchor_pts[2 * n], ay = anchor_pts[2 * n + 1];

  // B matrix (32x16 f16): lanes 0-15 hold K=0..15 (the real classes, CPAD=16),
  // lanes 16-31 hold K=16..31 (zero padding).  Same contiguous k order as A.
  v16h_t bmat;
#pragma unroll
  for (int i = 0; i < 16; ++i) bmat[i] = (_Float16)0.0f;
  if (lane < 16) bmat = *(const v16h_t*)(sig16 + (bN + n) * CPAD);

  // 4 x WMMA over the 64 GT rows: A = one-hot(labels) 16x32 f16.
  v8f_t cacc[4];
#pragma unroll
  for (int tq = 0; tq < 4; ++tq) {
    int m = tq * 16 + (lane & 15);
    int lbl = glb[m];
    v16h_t amat;
#pragma unroll
    for (int i = 0; i < 16; ++i) {
      int K = (lane < 16) ? i : i + 16;
      amat[i] = (K == lbl) ? (_Float16)1.0f : (_Float16)0.0f;
    }
    v8f_t z = {0.f, 0.f, 0.f, 0.f, 0.f, 0.f, 0.f, 0.f};
    cacc[tq] = __builtin_amdgcn_wmma_f32_16x16x32_f16(
        false, amat, false, bmat, (short)0, z, false, false);
  }

  int hf = lane >> 4;
  float bestIou = -1.0f, bestIouAlign = 0.0f;
  int bestIouM = M_;
  int cnt = 0, memM = M_;
  float memAlign = 0.0f, memIou = 0.0f;

#pragma unroll
  for (int tq = 0; tq < 4; ++tq) {
#pragma unroll
    for (int j = 0; j < 8; ++j) {
      int m = tq * 16 + j + 8 * hf;
      float score = cacc[tq][j];
      float gx = gbox[m][0], gy = gbox[m][1], gw = gbox[m][2], gh = gbox[m][3];
      float ca = gbox[m][4], sa = gbox[m][5];
      float hd = probiou_hd_dev(gx, gy, gbox[m][6], gbox[m][7], gbox[m][8],
                                px, py, pva, pvb, pvc);
      float iou = fmaxf(1.0f - hd, 0.0f);
      float i2 = iou * iou, i6 = i2 * i2 * i2;
      float align = score * i6;
      if (iou > bestIou) { bestIou = iou; bestIouM = m; bestIouAlign = align; }
      float dx = ax - gx, dy = ay - gy;
      float xr = dx * ca + dy * sa;
      float yr = -dx * sa + dy * ca;
      bool cand = (fabsf(xr) < gw * 0.5f) && (fabsf(yr) < gh * 0.5f) && (gval[m] > 0.0f);
      float key = cand ? align : 0.0f;
      if (key > EPSA && key >= gbox[m][9]) {
        ++cnt;
        if (m < memM) { memM = m; memAlign = align; memIou = iou; }
      }
    }
  }

  // merge lane pair (l, l+16): both hold same anchor column
  float oI = __shfl_xor(bestIou, 16);
  int   oM = __shfl_xor(bestIouM, 16);
  float oA = __shfl_xor(bestIouAlign, 16);
  if (oI > bestIou || (oI == bestIou && oM < bestIouM)) {
    bestIou = oI; bestIouM = oM; bestIouAlign = oA;
  }
  int   ocnt = __shfl_xor(cnt, 16);
  int   omM  = __shfl_xor(memM, 16);
  float omA  = __shfl_xor(memAlign, 16);
  float omI  = __shfl_xor(memIou, 16);
  if (omM < memM) { memM = omM; memAlign = omA; memIou = omI; }
  int fg0 = cnt + ocnt;

  int tgt; float fg, alT, ioT;
  if (fg0 > 1)       { tgt = bestIouM; alT = bestIouAlign; ioT = bestIou; fg = 1.0f; }
  else if (fg0 == 1) { tgt = memM;     alT = memAlign;     ioT = memIou;  fg = 1.0f; }
  else               { tgt = 0; alT = 0.0f; ioT = 0.0f; fg = 0.0f; }

  if (lane < 16) {
    tgtI[bN + n] = tgt;
    fgA[bN + n]  = fg;
    alA[bN + n]  = alT;
    ioA[bN + n]  = ioT;
    if (fg > 0.0f) {
      atomicMax(&maxAl[b * M_ + tgt], __float_as_uint(alT));
      atomicMax(&maxIo[b * M_ + tgt], __float_as_uint(ioT));
    }
  }
}

// ---------------------------------------------------------------- K4: per-anchor losses -> block partials
__global__ void k4_loss(const float* __restrict__ cls_logits,
                        const float* __restrict__ raw_angles,
                        const int* __restrict__ gt_labels,
                        const float* __restrict__ gt_boxes,
                        const float* __restrict__ dec,
                        const float* __restrict__ angstat,
                        const int* __restrict__ tgtI,
                        const float* __restrict__ fgA,
                        const float* __restrict__ alA,
                        const unsigned* __restrict__ maxAl,
                        const unsigned* __restrict__ maxIo,
                        float* __restrict__ partials, int N) {
  __shared__ float red[256];
  int tid = threadIdx.x;
  int idx = blockIdx.x * 256 + tid;
  float p[5] = {0.f, 0.f, 0.f, 0.f, 0.f};

  if (idx < B_ * N) {
    int b = idx / N;
    bool fg = fgA[idx] > 0.0f;
    int tgt = tgtI[idx];
    float na = 0.0f;
    int al = NC;
    if (fg) {
      float mA = __uint_as_float(maxAl[b * M_ + tgt]);
      float mI = __uint_as_float(maxIo[b * M_ + tgt]);
      na = alA[idx] * mI / (mA + EPSA);
      al = gt_labels[b * M_ + tgt];
    }
    const float* cl = cls_logits + (size_t)idx * NC;
    float lc = 0.0f;
    for (int c = 0; c < NC; ++c) {
      float x = cl[c];
      float sig = 1.0f / (1.0f + expf(-x));
      float tv = (c == al) ? na : 0.0f;
      float fw = (c == al) ? na : 0.75f * sig * sig;
      float bce = fmaxf(x, 0.0f) - x * tv + log1pf(expf(-fabsf(x)));
      lc += bce * fw;
    }
    p[0] = na;  // score_sum contribution
    p[1] = lc;
    if (fg) {
      const float* gb = gt_boxes + ((size_t)b * M_ + tgt) * 5;
      float gx = gb[0], gy = gb[1], gw = gb[2], gh = gb[3], ga = gb[4];
      float ca = cosf(ga), sa = sinf(ga);
      float Aq = gw * gw * (1.0f / 12.0f), Bq = gh * gh * (1.0f / 12.0f);
      float gva = Aq * ca * ca + Bq * sa * sa;
      float gvb = Aq * sa * sa + Bq * ca * ca;
      float gvc = (Aq - Bq) * ca * sa;
      const float4* dpv = (const float4*)(dec + (size_t)idx * 8);
      float4 d0 = dpv[0], d1 = dpv[1];
      float hd = probiou_hd_dev(d0.x, d0.y, d1.y, d1.z, d1.w, gx, gy, gva, gvb, gvc);
      p[2] = hd * na;

      float tt = fminf(fmaxf(ga / ANG_SCALE, 0.0f), 89.99f);
      int li = (int)tt;
      int ri = min(li + 1, 90);
      float lw = (float)ri - tt, rw = 1.0f - lw;
      float amax = angstat[(size_t)idx * 2 + 0];
      float lse  = angstat[(size_t)idx * 2 + 1];
      const float* ra = raw_angles + (size_t)idx * NLOG;
      float ce_l = -(ra[li] - amax - lse);
      float ce_r = -(ra[ri] - amax - lse);
      p[3] = ce_l * lw + ce_r * rw;
      p[4] = 1.0f;
    }
  }
  for (int c = 0; c < 5; ++c) {
    red[tid] = p[c];
    __syncthreads();
    for (int w = 128; w > 0; w >>= 1) {
      if (tid < w) red[tid] += red[tid + w];
      __syncthreads();
    }
    if (tid == 0) partials[(size_t)blockIdx.x * 5 + c] = red[0];
    __syncthreads();
  }
}

// ---------------------------------------------------------------- K5: deterministic final reduction
__global__ void k5_final(const float* __restrict__ partials, int NB,
                         float* __restrict__ out) {
  __shared__ float red[256];
  int tid = threadIdx.x;
  float s[5] = {0.f, 0.f, 0.f, 0.f, 0.f};
  for (int i = tid; i < NB; i += 256)
    for (int c = 0; c < 5; ++c) s[c] += partials[(size_t)i * 5 + c];
  float tot[5];
  for (int c = 0; c < 5; ++c) {
    red[tid] = s[c];
    __syncthreads();
    for (int w = 128; w > 0; w >>= 1) {
      if (tid < w) red[tid] += red[tid + w];
      __syncthreads();
    }
    tot[c] = red[0];
    __syncthreads();
  }
  if (tid == 0) {
    float ss   = fmaxf(tot[0], 1.0f);
    float lcls = tot[1] / ss;
    float lbox = tot[2] / ss;
    float npos = fmaxf(tot[4], 1.0f);
    float lang = tot[3] / npos;
    float total = 1.0f * lcls + 2.5f * lbox + 0.05f * lang;
    out[0] = total; out[1] = lcls; out[2] = lbox; out[3] = lang;
  }
}

// ---------------------------------------------------------------- launcher
extern "C" void kernel_launch(void* const* d_in, const int* in_sizes, int n_in,
                              void* d_out, int out_size, void* d_ws, size_t ws_size,
                              hipStream_t stream) {
  const float* cls_logits = (const float*)d_in[0];
  const float* reg_dist   = (const float*)d_in[1];
  const float* raw_angles = (const float*)d_in[2];
  const int*   gt_labels  = (const int*)d_in[3];
  const float* gt_boxes   = (const float*)d_in[4];
  const float* valid_mask = (const float*)d_in[5];
  const float* anchor_pts = (const float*)d_in[6];
  const float* stride_t   = (const float*)d_in[7];
  (void)n_in; (void)out_size; (void)ws_size;

  const int N  = in_sizes[0] / (B_ * NC);   // 21504
  const int BN = B_ * N;
  const int NB = (BN + 255) / 256;

  char* ws = (char*)d_ws;
  size_t off = 0;
  auto take = [&](size_t bytes) -> char* {
    char* p = ws + off;
    off = (off + bytes + 255) & ~(size_t)255;
    return p;
  };
  _Float16* sig16   = (_Float16*)take((size_t)BN * CPAD * 2);
  float*    dec     = (float*)take((size_t)BN * 8 * 4);
  float*    angstat = (float*)take((size_t)BN * 2 * 4);
  float*    thr     = (float*)take((size_t)B_ * M_ * 4);
  int*      tgtI    = (int*)take((size_t)BN * 4);
  float*    fgA     = (float*)take((size_t)BN * 4);
  float*    alA     = (float*)take((size_t)BN * 4);
  float*    ioA     = (float*)take((size_t)BN * 4);
  unsigned* maxAl   = (unsigned*)take((size_t)B_ * M_ * 4);
  unsigned* maxIo   = (unsigned*)take((size_t)B_ * M_ * 4);
  float*    parts   = (float*)take((size_t)NB * 5 * 4);

  k1_prep<<<(BN + 255) / 256, 256, 0, stream>>>(
      cls_logits, reg_dist, raw_angles, anchor_pts, stride_t,
      sig16, dec, angstat, N);
  k0_zero<<<(B_ * M_ + 255) / 256, 256, 0, stream>>>(maxAl, maxIo, B_ * M_);
  k2_topk<<<B_ * M_, 256, 0, stream>>>(
      dec, sig16, gt_labels, gt_boxes, valid_mask, anchor_pts, thr, N);
  k3_assign<<<B_ * (N / 16), 32, 0, stream>>>(
      dec, sig16, gt_labels, gt_boxes, valid_mask, anchor_pts, thr,
      tgtI, fgA, alA, ioA, maxAl, maxIo, N);
  k4_loss<<<NB, 256, 0, stream>>>(
      cls_logits, raw_angles, gt_labels, gt_boxes, dec, angstat,
      tgtI, fgA, alA, maxAl, maxIo, parts, N);
  k5_final<<<1, 256, 0, stream>>>(parts, NB, (float*)d_out);
}
